// GrowableTwistorLNN_24927990186623
// MI455X (gfx1250) — matrix operand order repaired and checked by reading, not compiled, not run on hardware
//
#include <hip/hip_runtime.h>
#include <hip/hip_bf16.h>
#include <math.h>

// Problem sizes (fixed by the reference)
#define TT 512
#define BB 64
#define DD 256
#define HH 1024
#define OO 64

typedef __bf16 bf16_t;
typedef __attribute__((ext_vector_type(16))) __bf16 v16bf;
typedef __attribute__((ext_vector_type(8)))  __bf16 v8bf;
typedef __attribute__((ext_vector_type(8)))  float  v8f;

// ---------------------------------------------------------------------------
// WMMA fragment loaders for V_WMMA_F32_16X16X32_BF16 (wave32).
//
// A (16x32, M x K), row-major source with leading dim `ld`:
//   lane<16 : M=lane,    K 0..15 ; lane>=16: M=lane-16, K 16..31
// -> per lane: two contiguous 16B chunks at k + hi*16 and k + hi*16 + 8.
//
// B (32x16, K x N) where B[k,n] = W[n,k] and W is row-major [N, K]:
//   lane<16 : N=lane,    K 0..15 ; lane>=16: N=lane-16, K 16..31
// -> per lane: 32 contiguous bytes of row W[n].
// ---------------------------------------------------------------------------
__device__ __forceinline__ v16bf load_a_frag(const bf16_t* base, int row, int ld,
                                             int kbase, int hi) {
  const bf16_t* p = base + (size_t)row * ld + kbase + hi * 16;
  v8bf lo = *(const v8bf*)(p);
  v8bf hh = *(const v8bf*)(p + 8);
  v16bf a;
#pragma unroll
  for (int i = 0; i < 8; ++i) { a[i] = lo[i]; a[i + 8] = hh[i]; }
  return a;
}

__device__ __forceinline__ v16bf load_b_frag(const bf16_t* base, int col, int ld,
                                             int kbase, int hi) {
  return *(const v16bf*)(base + (size_t)col * ld + kbase + hi * 16);
}

__device__ __forceinline__ float sigm(float x) { return 1.0f / (1.0f + __expf(-x)); }
__device__ __forceinline__ float clampf(float x, float lo, float hi) {
  return fminf(fmaxf(x, lo), hi);
}

#define WMMA_BF16(A, B, C) \
  __builtin_amdgcn_wmma_f32_16x16x32_bf16(false, (A), false, (B), (short)0, (C), false, false)

// ---------------------------------------------------------------------------
// Prep: fold sigmoid(mask) into recurrent weights, convert all GEMM operands
// to bf16 once (weights are time-invariant; x converted wholesale).
// ---------------------------------------------------------------------------
__global__ __launch_bounds__(256) void twistor_prep(
    const float* __restrict__ x, const float* __restrict__ W_real,
    const float* __restrict__ W_imag, const float* __restrict__ U_w,
    const float* __restrict__ W_tau_w, const float* __restrict__ mask_real,
    const float* __restrict__ mask_imag, const float* __restrict__ out_w,
    bf16_t* __restrict__ xb, bf16_t* __restrict__ Wr, bf16_t* __restrict__ Wi,
    bf16_t* __restrict__ Wt, bf16_t* __restrict__ Uw, bf16_t* __restrict__ Ow) {
  int i = blockIdx.x * 256 + threadIdx.x;
  const int NX = TT * BB * DD;           // 8,388,608 (largest range)
  if (i < NX) xb[i] = (bf16_t)x[i];
  if (i < HH * HH) {
    Wr[i] = (bf16_t)(W_real[i] * sigm(mask_real[i]));
    Wi[i] = (bf16_t)(W_imag[i] * sigm(mask_imag[i]));
    Wt[i] = (bf16_t)W_tau_w[i];
  }
  if (i < HH * DD) Uw[i] = (bf16_t)U_w[i];
  if (i < OO * HH) Ow[i] = (bf16_t)out_w[i];
}

// Zero-init f32 state and the t=0 bf16 operand buffer (z0 = 0 => tanh=0, |z|~0).
__global__ __launch_bounds__(256) void twistor_init(
    float* zr, float* zi, bf16_t* thr, bf16_t* thi, bf16_t* zm, bf16_t* zrb) {
  int i = blockIdx.x * 256 + threadIdx.x;   // BB*HH = 65536 total
  zr[i] = 0.0f; zi[i] = 0.0f;
  thr[i] = (bf16_t)0.0f; thi[i] = (bf16_t)0.0f;
  zm[i]  = (bf16_t)0.0f; zrb[i] = (bf16_t)0.0f;
}

// ---------------------------------------------------------------------------
// One fused scan step (kernel t):
//   blocks 0..31 : 256 waves, one 16x16 (batch, hidden) tile each.
//     4 WMMA accumulators: tanh(zr)@Wr^T, tanh(zi)@Wi^T, |z|@Wtau^T (K=1024)
//     and x_t@U_w^T (K=256). Then the fused liquid-ODE pointwise update,
//     writing f32 state in place and next-step bf16 operands (double-buffered).
//   blocks 32..33: pipelined readout of the *input* buffer's zr (state z_t,
//     produced by kernel t-1) -> y[t-1]. Read-only on the in-buffer, so no
//     hazard with the update workgroups writing the out-buffer.
// ---------------------------------------------------------------------------
__global__ __launch_bounds__(256) void twistor_step(
    int t,
    const bf16_t* __restrict__ xb,                              // [T,B,D]
    const bf16_t* __restrict__ Wr, const bf16_t* __restrict__ Wi,
    const bf16_t* __restrict__ Wt,                              // [H,H]
    const bf16_t* __restrict__ Uw,                              // [H,D]
    const bf16_t* __restrict__ Ow,                              // [O,H]
    const float* __restrict__ U_b, const float* __restrict__ W_tau_b,
    const float* __restrict__ tau_bias, const float* __restrict__ b_real,
    const float* __restrict__ b_imag, const float* __restrict__ out_b,
    float* __restrict__ zr, float* __restrict__ zi,             // [B,H] in-place
    const bf16_t* __restrict__ thr_in, const bf16_t* __restrict__ thi_in,
    const bf16_t* __restrict__ zm_in,  const bf16_t* __restrict__ zrb_in,
    bf16_t* __restrict__ thr_out, bf16_t* __restrict__ thi_out,
    bf16_t* __restrict__ zm_out,  bf16_t* __restrict__ zrb_out,
    float* __restrict__ y)                                      // [T,B,O]
{
  const int wave = threadIdx.x >> 5;
  const int lane = threadIdx.x & 31;
  const int hi   = lane >> 4;       // lane half selects K split / M half
  const int row  = lane & 15;

  if (blockIdx.x < 32) {
    // ----- state update tile -----
    const int gw = blockIdx.x * 8 + wave;        // 0..255
    const int m0 = (gw >> 6) * 16;               // batch tile (4)
    const int n0 = (gw & 63) * 16;               // hidden tile (64)
    v8f acc_r = {}, acc_i = {}, acc_t = {}, acc_u = {};

    for (int kb = 0; kb < HH; kb += 32) {
      v16bf ar = load_a_frag(thr_in, m0 + row, HH, kb, hi);
      v16bf br = load_b_frag(Wr,     n0 + row, HH, kb, hi);
      acc_r = WMMA_BF16(ar, br, acc_r);
      v16bf ai = load_a_frag(thi_in, m0 + row, HH, kb, hi);
      v16bf bi = load_b_frag(Wi,     n0 + row, HH, kb, hi);
      acc_i = WMMA_BF16(ai, bi, acc_i);
      v16bf am = load_a_frag(zm_in,  m0 + row, HH, kb, hi);
      v16bf bt = load_b_frag(Wt,     n0 + row, HH, kb, hi);
      acc_t = WMMA_BF16(am, bt, acc_t);
    }
    const bf16_t* xt = xb + (size_t)t * BB * DD;
    for (int kb = 0; kb < DD; kb += 32) {
      v16bf ax = load_a_frag(xt, m0 + row, DD, kb, hi);
      v16bf bu = load_b_frag(Uw, n0 + row, DD, kb, hi);
      acc_u = WMMA_BF16(ax, bu, acc_u);
    }

    const int h = n0 + row;
    const float brl = b_real[h], bim = b_imag[h], ub = U_b[h];
    const float wtb = W_tau_b[h], tbi = tau_bias[h];
#pragma unroll
    for (int r = 0; r < 8; ++r) {
      const int b   = m0 + r + hi * 8;           // C/D layout: m = vgpr + half*8
      const int idx = b * HH + h;
      const float zro = zr[idx], zio = zi[idx];
      const float ux  = acc_u[r] + ub;
      float dzr = -zro + acc_r[r] + ux + brl;
      float dzi = -zio + acc_i[r] + ux + bim;
      float tau = sigm(acc_t[r] + wtb) + tbi;
      tau = clampf(tau, 0.01f, 1.0f) + 1e-6f;
      dzr = clampf(dzr / tau, -10.0f, 10.0f);
      dzi = clampf(dzi / tau, -10.0f, 10.0f);
      const float zrn = clampf(zro + 0.1f * dzr, -100.0f, 100.0f);
      const float zin = clampf(zio + 0.1f * dzi, -100.0f, 100.0f);
      zr[idx] = zrn;  zi[idx] = zin;
      thr_out[idx] = (bf16_t)tanhf(zrn);
      thi_out[idx] = (bf16_t)tanhf(zin);
      zm_out[idx]  = (bf16_t)sqrtf(zrn * zrn + zin * zin + 1e-20f);
      zrb_out[idx] = (bf16_t)zrn;
    }
  } else {
    // ----- pipelined readout: y[t-1] = z_t(real) @ out_w^T + out_b -----
    if (t == 0) return;                           // no y for t = -1
    const int gw = (blockIdx.x - 32) * 8 + wave;  // 0..15
    const int m0 = (gw >> 2) * 16;
    const int n0 = (gw & 3) * 16;
    v8f acc = {};
    for (int kb = 0; kb < HH; kb += 32) {
      v16bf a  = load_a_frag(zrb_in, m0 + row, HH, kb, hi);
      v16bf bw = load_b_frag(Ow,     n0 + row, HH, kb, hi);
      acc = WMMA_BF16(a, bw, acc);
    }
    const int o = n0 + row;
    const float ob = out_b[o];
#pragma unroll
    for (int r = 0; r < 8; ++r) {
      const int b = m0 + r + hi * 8;
      y[(size_t)(t - 1) * BB * OO + (size_t)b * OO + o] = acc[r] + ob;
    }
  }
}

// Final readout for y[T-1] from the last state buffer.
__global__ __launch_bounds__(256) void twistor_readout(
    const bf16_t* __restrict__ zrb, const bf16_t* __restrict__ Ow,
    const float* __restrict__ out_b, float* __restrict__ y, int tOut) {
  const int wave = threadIdx.x >> 5;
  const int lane = threadIdx.x & 31;
  const int hi = lane >> 4, row = lane & 15;
  const int gw = blockIdx.x * 8 + wave;           // 0..15
  const int m0 = (gw >> 2) * 16;
  const int n0 = (gw & 3) * 16;
  v8f acc = {};
  for (int kb = 0; kb < HH; kb += 32) {
    v16bf a  = load_a_frag(zrb, m0 + row, HH, kb, hi);
    v16bf bw = load_b_frag(Ow,  n0 + row, HH, kb, hi);
    acc = WMMA_BF16(a, bw, acc);
  }
  const int o = n0 + row;
  const float ob = out_b[o];
#pragma unroll
  for (int r = 0; r < 8; ++r) {
    const int b = m0 + r + hi * 8;
    y[(size_t)tOut * BB * OO + (size_t)b * OO + o] = acc[r] + ob;
  }
}

// ---------------------------------------------------------------------------
// Launcher: prep + init, then 512 fused step kernels, then final readout.
// ---------------------------------------------------------------------------
extern "C" void kernel_launch(void* const* d_in, const int* in_sizes, int n_in,
                              void* d_out, int out_size, void* d_ws, size_t ws_size,
                              hipStream_t stream) {
  (void)in_sizes; (void)n_in; (void)out_size; (void)ws_size;
  const float* x        = (const float*)d_in[0];
  const float* W_real   = (const float*)d_in[1];
  const float* W_imag   = (const float*)d_in[2];
  const float* U_w      = (const float*)d_in[3];
  const float* U_b      = (const float*)d_in[4];
  const float* W_tau_w  = (const float*)d_in[5];
  const float* W_tau_b  = (const float*)d_in[6];
  const float* mask_real= (const float*)d_in[7];
  const float* mask_imag= (const float*)d_in[8];
  const float* tau_bias = (const float*)d_in[9];
  const float* b_real   = (const float*)d_in[10];
  const float* b_imag   = (const float*)d_in[11];
  const float* out_w    = (const float*)d_in[12];
  const float* out_b    = (const float*)d_in[13];
  float* y = (float*)d_out;

  // ---- workspace carve (~24 MB total) ----
  char* p = (char*)d_ws;
  auto carve = [&](size_t bytes) -> void* {
    void* r = (void*)p;
    p += (bytes + 255) & ~(size_t)255;
    return r;
  };
  bf16_t* xb  = (bf16_t*)carve((size_t)TT * BB * DD * 2);   // 16 MB
  bf16_t* Wr  = (bf16_t*)carve((size_t)HH * HH * 2);        // 2 MB
  bf16_t* Wi  = (bf16_t*)carve((size_t)HH * HH * 2);        // 2 MB
  bf16_t* Wt  = (bf16_t*)carve((size_t)HH * HH * 2);        // 2 MB
  bf16_t* Uw  = (bf16_t*)carve((size_t)HH * DD * 2);        // 0.5 MB
  bf16_t* Ow  = (bf16_t*)carve((size_t)OO * HH * 2);        // 128 KB
  float*  zr  = (float*)carve((size_t)BB * HH * 4);         // 256 KB
  float*  zi  = (float*)carve((size_t)BB * HH * 4);         // 256 KB
  bf16_t* thr[2], *thi[2], *zm[2], *zrb[2];
  for (int s = 0; s < 2; ++s) {
    thr[s] = (bf16_t*)carve((size_t)BB * HH * 2);
    thi[s] = (bf16_t*)carve((size_t)BB * HH * 2);
    zm[s]  = (bf16_t*)carve((size_t)BB * HH * 2);
    zrb[s] = (bf16_t*)carve((size_t)BB * HH * 2);
  }

  // ---- one-time operand prep (per call, deterministic) ----
  {
    const int NX = TT * BB * DD;
    twistor_prep<<<(NX + 255) / 256, 256, 0, stream>>>(
        x, W_real, W_imag, U_w, W_tau_w, mask_real, mask_imag, out_w,
        xb, Wr, Wi, Wt, Uw, Ow);
    twistor_init<<<(BB * HH) / 256, 256, 0, stream>>>(
        zr, zi, thr[0], thi[0], zm[0], zrb[0]);
  }

  // ---- serial scan: one fused kernel per timestep (graph-friendly) ----
  for (int t = 0; t < TT; ++t) {
    const int in_s  = t & 1;
    const int out_s = (t + 1) & 1;
    twistor_step<<<34, 256, 0, stream>>>(
        t, xb, Wr, Wi, Wt, Uw, Ow,
        U_b, W_tau_b, tau_bias, b_real, b_imag, out_b,
        zr, zi,
        thr[in_s],  thi[in_s],  zm[in_s],  zrb[in_s],
        thr[out_s], thi[out_s], zm[out_s], zrb[out_s],
        y);
  }
  // z_512 lives in buffer (TT & 1) == 0
  twistor_readout<<<2, 256, 0, stream>>>(zrb[0], Ow, out_b, y, TT - 1);
}
